// RNN_55405078119028
// MI455X (gfx1250) — matrix operand order repaired and checked by reading, not compiled, not run on hardware
//
#include <hip/hip_runtime.h>

// ---------------------------------------------------------------------------
// Leaky tanh-RNN scan for MI455X (gfx1250, wave32, WMMA).
//   h_t = (1-r) * tanh(x_t@i2h + h@h2h + k*a_t@a2h + reward@r2h + bh) + r*h
// Strategy:
//  * rbias = reward@r2h + bh precomputed once (C-accumulator init).
//  * Augmented weight W = [h2h ; i2h ; k*a2h ; 0]  (K = 1056) pre-packed in
//    bf16 into the exact V_WMMA_F32_16X16X32_BF16 B-fragment register layout.
//  * Hidden state kept (ping-pong) in bf16 pre-packed A-fragment layout, so
//    the per-step inner loop is: 3x coalesced 32B/lane loads + 2x v_wmma.
//  * One kernel launch per timestep sequences the recurrence on `stream`.
// ---------------------------------------------------------------------------

#define T_STEPS 256
#define B_SZ    256
#define I_SZ    9
#define H_SZ    1024
#define A_SZ    4
#define R_SZ    38
#define NMT     (B_SZ / 16)   // 16 row tiles
#define NNT     (H_SZ / 16)   // 64 col tiles
#define KCH     33            // 32 hidden K-chunks of 32 + 1 augmented chunk

typedef __attribute__((ext_vector_type(16))) __bf16 v16bf;
typedef __attribute__((ext_vector_type(8)))  float  v8f;

__device__ __forceinline__ unsigned short f32_to_bf16_rne(float f) {
  unsigned int u = __float_as_uint(f);
  u += 0x7FFFu + ((u >> 16) & 1u);          // round-to-nearest-even
  return (unsigned short)(u >> 16);
}

// Scatter one value of hidden[b][j] into the A-fragment-packed bf16 buffer.
// A 16-bit 16x32 fragment: lane = half*16 + row, per-lane element e maps to
// k = (e>>3)*16 + half*8 + (e&7).  Block id = chunk*NMT + (b/16).
__device__ __forceinline__ void pack_store_a(unsigned short* __restrict__ Ap,
                                             int b, int j, float v) {
  int c    = j >> 5;
  int rl   = j & 31;
  int e    = ((rl >> 4) << 3) | (rl & 7);
  int lane = (((rl >> 3) & 1) << 4) | (b & 15);
  Ap[(size_t)(c * NMT + (b >> 4)) * 512 + lane * 16 + e] = f32_to_bf16_rne(v);
}

// ---------------- one-time prep kernels ------------------------------------

// Pack augmented weight [h2h ; i2h ; k*a2h ; 0] into B-fragment layout:
// lane = half*16 + (n%16) holds K = kc + (e>>3)*16 + half*8 + (e&7)  for column n.
__global__ void pack_w_kernel(unsigned short* __restrict__ Wpack,
                              const float* __restrict__ h2h,
                              const float* __restrict__ i2h,
                              const float* __restrict__ a2h,
                              const int* __restrict__ kact) {
  int idx  = blockIdx.x * blockDim.x + threadIdx.x;
  int e    = idx & 15;
  int lane = (idx >> 4) & 31;
  int blk  = idx >> 9;
  int nt   = blk % NNT;
  int c    = blk / NNT;
  int k = c * 32 + ((e >> 3) << 4) + ((lane >> 4) << 3) + (e & 7);
  int n = nt * 16 + (lane & 15);
  float val = 0.f;
  if (k < H_SZ) {
    val = h2h[(size_t)k * H_SZ + n];
  } else {
    int kr = k - H_SZ;
    if (kr < I_SZ)             val = i2h[(size_t)kr * H_SZ + n];
    else if (kr < I_SZ + A_SZ) val = (float)kact[0] * a2h[(size_t)(kr - I_SZ) * H_SZ + n];
  }
  Wpack[idx] = f32_to_bf16_rne(val);
}

__global__ void rbias_kernel(float* __restrict__ rbias,
                             const float* __restrict__ reward,
                             const float* __restrict__ r2h,
                             const float* __restrict__ bh) {
  int idx = blockIdx.x * blockDim.x + threadIdx.x;
  int b = idx >> 10;
  int n = idx & (H_SZ - 1);
  float s = bh[n];
#pragma unroll
  for (int rr = 0; rr < R_SZ; ++rr)
    s += reward[b * R_SZ + rr] * r2h[rr * H_SZ + n];
  rbias[idx] = s;
}

__global__ void pack_h0_kernel(unsigned short* __restrict__ Ap,
                               const float* __restrict__ h0) {
  int idx = blockIdx.x * blockDim.x + threadIdx.x;
  pack_store_a(Ap, idx >> 10, idx & (H_SZ - 1), h0[idx]);
}

// ---------------- per-timestep GEMM + tanh + leaky blend -------------------

__global__ void __launch_bounds__(256)
rnn_step_kernel(const unsigned short* __restrict__ Wpack,
                const float* __restrict__ rbias,
                const unsigned short* __restrict__ Aprev,
                unsigned short* __restrict__ Anext,
                const float* __restrict__ hprev,   // [B,H] previous hidden f32
                const float* __restrict__ inp_t,   // [B,I]
                const float* __restrict__ act_t,   // [B,A]
                const float* __restrict__ rvec,    // [H] inertia
                float* __restrict__ out_t) {       // [B,H]
  const int tid  = threadIdx.x;
  const int lane = tid & 31;
  const int wid  = tid >> 5;
  const int half = lane >> 4;
  // 8 waves: 2 (M) x 4 (N); wave tile = 32 rows x 16 cols
  const int n0   = blockIdx.x * 64 + (wid >> 1) * 16;
  const int m0   = blockIdx.y * 64 + (wid & 1) * 32;
  const int mt0  = m0 >> 4;
  const int nt   = n0 >> 4;
  const int ncol = n0 + (lane & 15);

  // C/D layout: VGPR v -> row m0 + v + half*8, col = n0 + lane%16
  v8f acc0, acc1;
#pragma unroll
  for (int v = 0; v < 8; ++v) {
    acc0[v] = rbias[(size_t)(m0 + v + half * 8) * H_SZ + ncol];
    acc1[v] = rbias[(size_t)(m0 + 16 + v + half * 8) * H_SZ + ncol];
  }

  const unsigned short* Ab = Aprev + (size_t)mt0 * 512 + lane * 16;
  const unsigned short* Bb = Wpack + (size_t)nt  * 512 + lane * 16;

#pragma unroll 4
  for (int c = 0; c < 32; ++c) {
    v16bf a0 = *(const v16bf*)(Ab + (size_t)c * (NMT * 512));
    v16bf a1 = *(const v16bf*)(Ab + (size_t)c * (NMT * 512) + 512);
    v16bf bm = *(const v16bf*)(Bb + (size_t)c * (NNT * 512));
    acc0 = __builtin_amdgcn_wmma_f32_16x16x32_bf16(false, a0, false, bm,
                                                   (short)0, acc0, false, false);
    acc1 = __builtin_amdgcn_wmma_f32_16x16x32_bf16(false, a1, false, bm,
                                                   (short)0, acc1, false, false);
  }

  // Augmented chunk: columns [inp(9) | act(4) | zeros], gathered on the fly.
  {
    union { v16bf v; unsigned short s[16]; } ua0, ua1;
    const int row = lane & 15;
#pragma unroll
    for (int e = 0; e < 16; ++e) {
      int rl = ((e >> 3) << 4) + half * 8 + (e & 7);   // rel K in 0..31
      float v0 = 0.f, v1 = 0.f;
      if (rl < I_SZ) {
        v0 = inp_t[(m0 + row) * I_SZ + rl];
        v1 = inp_t[(m0 + 16 + row) * I_SZ + rl];
      } else if (rl < I_SZ + A_SZ) {
        v0 = act_t[(m0 + row) * A_SZ + (rl - I_SZ)];
        v1 = act_t[(m0 + 16 + row) * A_SZ + (rl - I_SZ)];
      }
      ua0.s[e] = f32_to_bf16_rne(v0);
      ua1.s[e] = f32_to_bf16_rne(v1);
    }
    v16bf bm = *(const v16bf*)(Bb + (size_t)32 * (NNT * 512));
    acc0 = __builtin_amdgcn_wmma_f32_16x16x32_bf16(false, ua0.v, false, bm,
                                                   (short)0, acc0, false, false);
    acc1 = __builtin_amdgcn_wmma_f32_16x16x32_bf16(false, ua1.v, false, bm,
                                                   (short)0, acc1, false, false);
  }

  // Epilogue: tanh + leaky blend, write f32 out + bf16 packed hidden for t+1.
  const float rj = rvec[ncol];
#pragma unroll
  for (int v = 0; v < 8; ++v) {
    {
      int rowg = m0 + v + half * 8;
      float hp = hprev[(size_t)rowg * H_SZ + ncol];
      float hn = (1.f - rj) * tanhf(acc0[v]) + rj * hp;
      out_t[(size_t)rowg * H_SZ + ncol] = hn;
      pack_store_a(Anext, rowg, ncol, hn);
    }
    {
      int rowg = m0 + 16 + v + half * 8;
      float hp = hprev[(size_t)rowg * H_SZ + ncol];
      float hn = (1.f - rj) * tanhf(acc1[v]) + rj * hp;
      out_t[(size_t)rowg * H_SZ + ncol] = hn;
      pack_store_a(Anext, rowg, ncol, hn);
    }
  }
}

// ---------------- host-side launch -----------------------------------------

extern "C" void kernel_launch(void* const* d_in, const int* in_sizes, int n_in,
                              void* d_out, int out_size, void* d_ws, size_t ws_size,
                              hipStream_t stream) {
  (void)in_sizes; (void)n_in; (void)out_size; (void)ws_size;

  const float* inputs  = (const float*)d_in[0];   // [T,B,I]
  const float* hidden0 = (const float*)d_in[1];   // [B,H]
  const float* actions = (const float*)d_in[2];   // [T,B,A]
  const float* reward  = (const float*)d_in[3];   // [B,R]
  const float* h2h_i   = (const float*)d_in[4];   // i2h [I,H]
  const float* h2h     = (const float*)d_in[5];   // h2h [H,H]
  const float* a2h     = (const float*)d_in[6];   // a2h [A,H]
  const float* r2h     = (const float*)d_in[7];   // r2h [R,H]
  const float* bh      = (const float*)d_in[8];   // [1,H]
  const float* rvec    = (const float*)d_in[9];   // [1,H]
  const int*   kact    = (const int*)d_in[10];    // scalar

  // workspace carve-up (~4.1 MB total)
  constexpr size_t WPACK_BYTES = (size_t)KCH * NNT * 512 * 2;   // 2,162,688
  constexpr size_t RBIAS_BYTES = (size_t)B_SZ * H_SZ * 4;       // 1,048,576
  constexpr size_t APACK_BYTES = (size_t)B_SZ * H_SZ * 2;       //   524,288
  char* ws = (char*)d_ws;
  unsigned short* Wpack = (unsigned short*)ws;
  float*          rbias = (float*)(ws + WPACK_BYTES);
  unsigned short* Ap0   = (unsigned short*)(ws + WPACK_BYTES + RBIAS_BYTES);
  unsigned short* Ap1   = (unsigned short*)(ws + WPACK_BYTES + RBIAS_BYTES + APACK_BYTES);

  const int WTOT = KCH * NNT * 512;                       // 1,081,344
  pack_w_kernel<<<WTOT / 256, 256, 0, stream>>>(Wpack, h2h, h2h_i, a2h, kact);
  rbias_kernel<<<(B_SZ * H_SZ) / 256, 256, 0, stream>>>(rbias, reward, r2h, bh);
  pack_h0_kernel<<<(B_SZ * H_SZ) / 256, 256, 0, stream>>>(Ap0, hidden0);

  float* out = (float*)d_out;
  dim3 grid(H_SZ / 64, B_SZ / 64);   // 16 x 4
  for (int t = 0; t < T_STEPS; ++t) {
    const unsigned short* Aprev = (t & 1) ? Ap1 : Ap0;
    unsigned short*       Anext = (t & 1) ? Ap0 : Ap1;
    const float* hprev = (t == 0) ? hidden0 : out + (size_t)(t - 1) * B_SZ * H_SZ;
    rnn_step_kernel<<<grid, 256, 0, stream>>>(
        Wpack, rbias, Aprev, Anext, hprev,
        inputs + (size_t)t * B_SZ * I_SZ,
        actions + (size_t)t * B_SZ * A_SZ,
        rvec,
        out + (size_t)t * B_SZ * H_SZ);
  }
}
